// unit_gtcn_T_32014686224473
// MI455X (gfx1250) — matrix operand order repaired and checked by reading, not compiled, not run on hardware
//
#include <hip/hip_runtime.h>
#include <math.h>

typedef __bf16 bf16_t;
typedef bf16_t v16bf __attribute__((ext_vector_type(16)));
typedef bf16_t v8bf  __attribute__((ext_vector_type(8)));
typedef float  v8f   __attribute__((ext_vector_type(8)));

#define N_   64
#define C_   64
#define T_   300
#define V_   25
#define IC_  16
#define S_   3
#define TP_  312        // padded T: 4 left + 300 + 8 right
#define KCONV (C_*9)    // 576, K order: k = kt*64 + c
#define KATT  (IC_*T_)  // 4800
#define P_    (T_*V_)   // 7500

static __device__ __forceinline__ bf16_t f2bf(float f) {
  union { float f; unsigned u; } a; a.f = f;
  unsigned r = a.u + 0x7FFFu + ((a.u >> 16) & 1u);
  union { unsigned short s; bf16_t b; } o; o.s = (unsigned short)(r >> 16);
  return o.b;
}

// A-fragment: lane<16 holds K kb..kb+15, lane>=16 holds K kb+16..kb+31 (matches B)
static __device__ __forceinline__ v16bf ldA(const bf16_t* __restrict__ row, int kb, int lane) {
  int kA = kb + ((lane & 16) ? 16 : 0);
  v8bf lo = *(const v8bf*)(row + kA);
  v8bf hi = *(const v8bf*)(row + kA + 8);
  v16bf r;
#pragma unroll
  for (int h = 0; h < 8; ++h) { r[h] = lo[h]; r[8 + h] = hi[h]; }
  return r;
}

// B-fragment: lane<16 holds K kb..kb+15, lane>=16 holds K kb+16..kb+31 (matches A)
static __device__ __forceinline__ v16bf ldB(const bf16_t* __restrict__ row, int kb, int lane) {
  int kB = kb + ((lane & 16) ? 16 : 0);
  v8bf lo = *(const v8bf*)(row + kB);
  v8bf hi = *(const v8bf*)(row + kB + 8);
  v16bf r;
#pragma unroll
  for (int h = 0; h < 8; ++h) { r[h] = lo[h]; r[8 + h] = hi[h]; }
  return r;
}

static __device__ __forceinline__ float slim16(int o, const float* w) {
  if (o < 9)  return 1.0f;
  if (o < 11) return w[1];
  if (o < 12) return w[2];
  if (o < 14) return w[3];
  return w[4];
}
static __device__ __forceinline__ float slim64(int o, const float* w) {
  if (o < 38) return 1.0f;
  if (o < 44) return w[1];
  if (o < 51) return w[2];
  if (o < 57) return w[3];
  return w[4];
}

// ---------------- prep kernels ----------------

__global__ void k_prep_afull(const float* __restrict__ A, const float* __restrict__ PA,
                             float* __restrict__ afull, int tot) {
  int i = blockIdx.x * blockDim.x + threadIdx.x;
  if (i >= tot) return;
  int rem = i % (V_ * V_);
  int w = rem / V_, v = rem % V_;
  float a = A[i];
  float a2 = a * a, a4 = a2 * a2;
  float ach = 8.0f * a4 - 4.0f * a2 - 4.0f * a + ((w == v) ? 1.0f : 0.0f);
  afull[i] = a + PA[i] + ach;
}

// conv weights permuted to kt-major K: w1p[(s*16+ic)*576 + kt*64 + c]
__global__ void k_prep_w(const float* __restrict__ WT1, const float* __restrict__ WT2,
                         const float* __restrict__ Wd,
                         bf16_t* __restrict__ w1p, bf16_t* __restrict__ w2p,
                         bf16_t* __restrict__ wdb) {
  const int n1 = S_ * IC_ * KCONV;          // 27648
  const int n2 = 2 * n1;                    // 55296
  const int nt = n2 + S_ * C_ * C_;         // 67584
  int i = blockIdx.x * blockDim.x + threadIdx.x;
  if (i >= nt) return;
  if (i < n2) {
    int j = (i < n1) ? i : (i - n1);
    int kt  = j % 9;
    int c   = (j / 9) % C_;
    int ics = j / (9 * C_);                 // s*16+ic
    int dst = ics * KCONV + kt * C_ + c;
    if (i < n1) w1p[dst] = f2bf(WT1[j]);
    else        w2p[dst] = f2bf(WT2[j]);
  } else {
    wdb[i - n2] = f2bf(Wd[i - n2]);
  }
}

// xbp[n][v][tp][c] bf16 (c-contiguous 128B rows), tp = t+4, zero pad tp<4 and tp>=304
__global__ void k_prep_x(const float* __restrict__ x, bf16_t* __restrict__ xbp) {
  size_t tot = (size_t)N_ * V_ * TP_ * C_;
  size_t i = (size_t)blockIdx.x * blockDim.x + threadIdx.x;
  if (i >= tot) return;
  int c  = (int)(i % C_);
  int tp = (int)((i / C_) % TP_);
  int v  = (int)((i / ((size_t)C_ * TP_)) % V_);
  int n  = (int)(i / ((size_t)C_ * TP_ * V_));
  float val = 0.0f;
  int t = tp - 4;
  if (t >= 0 && t < T_)
    val = x[((size_t)(n * C_ + c) * T_ + t) * V_ + v];
  xbp[i] = f2bf(val);
}

// ---------------- temporal conv: t1,t2 = slim_conv(x, WT1/WT2) ----------------
// one wave per (n, v, t-block of 16). M=ic(16), N=t(16), K=kt*64+c (576) = 18 wmma steps.
// Window 24x64 bf16 = 3KB contiguous in global -> async global->LDS copy.
__global__ void k_tconv(const bf16_t* __restrict__ xbp,
                        const bf16_t* __restrict__ w1p, const bf16_t* __restrict__ w2p,
                        const float* __restrict__ bT1, const float* __restrict__ bT2,
                        const float* __restrict__ wvec,
                        bf16_t* __restrict__ t1b, bf16_t* __restrict__ t2b, int s) {
  __shared__ __align__(16) bf16_t xwin[24 * C_];   // [j=0..23][c]
  const int lane = threadIdx.x;
  const int n = blockIdx.x, v = blockIdx.y, tb = blockIdx.z;
  const int t0 = tb * 16;

  // async-stage the contiguous 3KB window: 192 x 16B chunks, 6 per lane
  {
    const char* gsrc = (const char*)(xbp + ((size_t)(n * V_ + v) * TP_ + t0) * C_);
    unsigned ldsbase = (unsigned)(uintptr_t)(void*)xwin;   // addr[31:0] == LDS offset
#pragma unroll
    for (int i = 0; i < 6; ++i) {
      int q = lane + i * 32;
      unsigned dst = ldsbase + (unsigned)(q * 16);
      const char* src = gsrc + q * 16;
      asm volatile("global_load_async_to_lds_b128 %0, %1, off"
                   :: "v"(dst), "v"(src) : "memory");
    }
    asm volatile("s_wait_asynccnt 0" ::: "memory");
  }
  __syncthreads();

  const bf16_t* a1row = w1p + ((size_t)(s * IC_) + (lane & 15)) * KCONV;
  const bf16_t* a2row = w2p + ((size_t)(s * IC_) + (lane & 15)) * KCONV;
  const int tt = lane & 15;
  const int bbase = (lane & 16) ? 16 : 0;

  v8f acc1 = {}; v8f acc2 = {};
  for (int ks = 0; ks < KCONV / 32; ++ks) {
    int kb = ks * 32;
    v16bf a1 = ldA(a1row, kb, lane);
    v16bf a2 = ldA(a2row, kb, lane);
    // B fragment: 16 consecutive k share one kt (kB&63 in {0,16,32,48})
    int kB = kb + bbase;
    int kt = kB >> 6, c0 = kB & 63;
    const bf16_t* lp = xwin + (tt + kt) * C_ + c0;
    v8bf lo = *(const v8bf*)lp;
    v8bf hi = *(const v8bf*)(lp + 8);
    v16bf b;
#pragma unroll
    for (int h = 0; h < 8; ++h) { b[h] = lo[h]; b[8 + h] = hi[h]; }
    acc1 = __builtin_amdgcn_wmma_f32_16x16x32_bf16(false, a1, false, b, (short)0, acc1, false, false);
    acc2 = __builtin_amdgcn_wmma_f32_16x16x32_bf16(false, a2, false, b, (short)0, acc2, false, false);
  }

  const int t = t0 + tt;
  if (t < T_) {
#pragma unroll
    for (int r = 0; r < 8; ++r) {
      int ic = r + ((lane & 16) ? 8 : 0);
      size_t o = ((size_t)(n * 32 + v) * IC_ + ic) * T_ + t;  // [n][v_pad32][ic*300+t]
      t1b[o] = f2bf((acc1[r] + bT1[s * IC_ + ic]) * slim16(ic, wvec));
      t2b[o] = f2bf((acc2[r] + bT2[s * IC_ + ic]) * slim16(ic, wvec));
    }
  }
}

// ---------------- attention logits: att[n][vq][vk] = sum_K t1T*t2T ----------------
__global__ void k_att(const bf16_t* __restrict__ t1b, const bf16_t* __restrict__ t2b,
                      float* __restrict__ att) {
  int n = blockIdx.x;
  int ti = blockIdx.y >> 1, tj = blockIdx.y & 1;
  int lane = threadIdx.x;
  const bf16_t* arow = t1b + ((size_t)n * 32 + ti * 16 + (lane & 15)) * KATT;
  const bf16_t* brow = t2b + ((size_t)n * 32 + tj * 16 + (lane & 15)) * KATT;
  v8f acc = {};
  for (int ks = 0; ks < KATT / 32; ++ks) {
    v16bf a = ldA(arow, ks * 32, lane);
    v16bf b = ldB(brow, ks * 32, lane);
    acc = __builtin_amdgcn_wmma_f32_16x16x32_bf16(false, a, false, b, (short)0, acc, false, false);
  }
#pragma unroll
  for (int r = 0; r < 8; ++r) {
    int row = ti * 16 + r + ((lane & 16) ? 8 : 0);
    int col = tj * 16 + (lane & 15);
    att[((size_t)n * 32 + row) * 32 + col] = acc[r];
  }
}

// ---------------- softmax over vq + A_full -> A1T[n][v][w] bf16 (K-pad zeroed) ----------------
__global__ void k_soft(const float* __restrict__ att, const float* __restrict__ afull,
                       bf16_t* __restrict__ a1t, int s) {
  int n = blockIdx.x, v = threadIdx.x;
  bf16_t* dst = a1t + ((size_t)n * 32 + v) * 32;
  if (v < V_) {
    const float sc = 1.0f / (float)KATT;
    float m = -1e30f;
    for (int w = 0; w < V_; ++w)
      m = fmaxf(m, att[((size_t)n * 32 + w) * 32 + v] * sc);
    float ssum = 0.0f;
    for (int w = 0; w < V_; ++w)
      ssum += __expf(att[((size_t)n * 32 + w) * 32 + v] * sc - m);
    float inv = 1.0f / ssum;
    for (int w = 0; w < V_; ++w)
      dst[w] = f2bf(__expf(att[((size_t)n * 32 + w) * 32 + v] * sc - m) * inv
                    + afull[(s * V_ + w) * V_ + v]);
    for (int w = V_; w < 32; ++w) dst[w] = f2bf(0.0f);
  } else {
    for (int w = 0; w < 32; ++w) dst[w] = f2bf(0.0f);
  }
}

// ---------------- agg = X @ A1 -> aggT[n][t*25+v][c] bf16 (K=25 padded to 32) ----------------
__global__ void k_agg(const float* __restrict__ x, const bf16_t* __restrict__ a1t,
                      bf16_t* __restrict__ aggT) {
  int lin = blockIdx.x;
  int n = lin / 19, tb = lin % 19;
  int c = blockIdx.y;
  int lane = threadIdx.x;
  int t0 = tb * 16;
  int tt = lane & 15;
  int tA = t0 + tt;
  bool tok = tA < T_;
  const float* rowx = x + ((size_t)(n * C_ + c) * T_ + (tok ? tA : 0)) * V_;
  int kApat = (lane & 16) ? 16 : 0;
  v16bf a;
#pragma unroll
  for (int h = 0; h < 8; ++h) {
    int k0 = kApat + h, k1 = kApat + 8 + h;
    a[h]     = (tok && k0 < V_) ? f2bf(rowx[k0]) : f2bf(0.0f);
    a[8 + h] = (tok && k1 < V_) ? f2bf(rowx[k1]) : f2bf(0.0f);
  }
  const bf16_t* b0row = a1t + ((size_t)n * 32 + (lane & 15)) * 32;
  const bf16_t* b1row = a1t + ((size_t)n * 32 + 16 + (lane & 15)) * 32;
  v16bf b0 = ldB(b0row, 0, lane);
  v16bf b1 = ldB(b1row, 0, lane);
  v8f acc0 = {}; v8f acc1 = {};
  acc0 = __builtin_amdgcn_wmma_f32_16x16x32_bf16(false, a, false, b0, (short)0, acc0, false, false);
  acc1 = __builtin_amdgcn_wmma_f32_16x16x32_bf16(false, a, false, b1, (short)0, acc1, false, false);
#pragma unroll
  for (int r = 0; r < 8; ++r) {
    int trow = t0 + r + ((lane & 16) ? 8 : 0);
    if (trow < T_) {
      int v0 = lane & 15;
      aggT[((size_t)n * P_ + trow * V_ + v0) * C_ + c] = f2bf(acc0[r]);
      int v1 = 16 + (lane & 15);
      if (v1 < V_)
        aggT[((size_t)n * P_ + trow * V_ + v1) * C_ + c] = f2bf(acc1[r]);
    }
  }
}

// ---------------- z = slim_conv1x1(agg, Wd) accumulated into out ----------------
__global__ void k_z(const bf16_t* __restrict__ wdb, const bf16_t* __restrict__ aggT,
                    const float* __restrict__ bd, const float* __restrict__ wvec,
                    float* __restrict__ out, int s) {
  int n = blockIdx.x, o0 = blockIdx.y * 16, chunk = blockIdx.z;
  int lane = threadIdx.x;
  const bf16_t* arow = wdb + ((size_t)(s * C_) + o0 + (lane & 15)) * C_;
  v16bf aK0 = ldA(arow, 0, lane);   // loop-invariant weights
  v16bf aK1 = ldA(arow, 32, lane);
  for (int it = 0; it < 10; ++it) {
    int p0 = (chunk * 10 + it) * 16;
    if (p0 >= P_) break;
    int pcol = p0 + (lane & 15);
    int pc = (pcol < P_) ? pcol : (P_ - 1);
    const bf16_t* brow = aggT + ((size_t)n * P_ + pc) * C_;
    v16bf b0 = ldB(brow, 0, lane);
    v16bf b1 = ldB(brow, 32, lane);
    v8f acc = {};
    acc = __builtin_amdgcn_wmma_f32_16x16x32_bf16(false, aK0, false, b0, (short)0, acc, false, false);
    acc = __builtin_amdgcn_wmma_f32_16x16x32_bf16(false, aK1, false, b1, (short)0, acc, false, false);
    if (pcol < P_) {
#pragma unroll
      for (int r = 0; r < 8; ++r) {
        int o = o0 + r + ((lane & 16) ? 8 : 0);
        float val = (acc[r] + bd[s * C_ + o]) * slim64(o, wvec);
        size_t idx = ((size_t)(n * C_) + o) * P_ + pcol;
        if (s == 0) out[idx] = val;
        else        out[idx] += val;
      }
    }
  }
}

// ---------------- BN + residual + relu ----------------
__global__ void k_bn(float* __restrict__ out, const float* __restrict__ x,
                     const float* __restrict__ gamma, const float* __restrict__ beta,
                     const float* __restrict__ mean, const float* __restrict__ var) {
  size_t tot = (size_t)N_ * C_ * P_;
  size_t i = (size_t)blockIdx.x * blockDim.x + threadIdx.x;
  if (i >= tot) return;
  int o = (int)((i / P_) % C_);
  float y = out[i];
  float inv = rsqrtf(var[o] + 1e-5f);
  y = (y - mean[o]) * (gamma[o] * inv) + beta[o] + x[i];
  out[i] = fmaxf(y, 0.0f);
}

extern "C" void kernel_launch(void* const* d_in, const int* in_sizes, int n_in,
                              void* d_out, int out_size, void* d_ws, size_t ws_size,
                              hipStream_t stream) {
  (void)in_sizes; (void)n_in; (void)out_size; (void)ws_size;
  const float* x     = (const float*)d_in[0];
  const float* wvec  = (const float*)d_in[1];
  const float* A     = (const float*)d_in[2];
  const float* PA    = (const float*)d_in[3];
  const float* Wd    = (const float*)d_in[4];
  const float* bd    = (const float*)d_in[5];
  const float* WT1   = (const float*)d_in[6];
  const float* bT1   = (const float*)d_in[7];
  const float* WT2   = (const float*)d_in[8];
  const float* bT2   = (const float*)d_in[9];
  const float* gamma = (const float*)d_in[10];
  const float* beta  = (const float*)d_in[11];
  const float* mean  = (const float*)d_in[12];
  const float* var   = (const float*)d_in[13];
  float* out = (float*)d_out;

  char* ws = (char*)d_ws;
  size_t off = 0;
  auto take = [&](size_t bytes) -> char* {
    char* p = ws + off;
    off += (bytes + 255) & ~(size_t)255;
    return p;
  };
  bf16_t* xbp   = (bf16_t*)take((size_t)N_ * V_ * TP_ * C_ * 2);   // ~63.9 MB
  bf16_t* w1p   = (bf16_t*)take((size_t)S_ * IC_ * KCONV * 2);
  bf16_t* w2p   = (bf16_t*)take((size_t)S_ * IC_ * KCONV * 2);
  bf16_t* wdb   = (bf16_t*)take((size_t)S_ * C_ * C_ * 2);
  float*  afull = (float*) take((size_t)S_ * V_ * V_ * 4);
  bf16_t* t1b   = (bf16_t*)take((size_t)N_ * 32 * KATT * 2);       // ~19.7 MB
  bf16_t* t2b   = (bf16_t*)take((size_t)N_ * 32 * KATT * 2);
  float*  att   = (float*) take((size_t)N_ * 32 * 32 * 4);
  bf16_t* a1t   = (bf16_t*)take((size_t)N_ * 32 * 32 * 2);
  bf16_t* aggT  = (bf16_t*)take((size_t)N_ * P_ * C_ * 2);         // ~61.4 MB

  {
    int tot = S_ * V_ * V_;
    k_prep_afull<<<(tot + 255) / 256, 256, 0, stream>>>(A, PA, afull, tot);
  }
  {
    int tot = 2 * S_ * IC_ * KCONV + S_ * C_ * C_;
    k_prep_w<<<(tot + 255) / 256, 256, 0, stream>>>(WT1, WT2, Wd, w1p, w2p, wdb);
  }
  {
    size_t tot = (size_t)N_ * V_ * TP_ * C_;
    k_prep_x<<<(unsigned)((tot + 255) / 256), 256, 0, stream>>>(x, xbp);
  }

  for (int s = 0; s < S_; ++s) {
    dim3 g1(N_, V_, 19);
    k_tconv<<<g1, 32, 0, stream>>>(xbp, w1p, w2p, bT1, bT2, wvec, t1b, t2b, s);
    dim3 g2(N_, 4);
    k_att<<<g2, 32, 0, stream>>>(t1b, t2b, att);
    k_soft<<<N_, 32, 0, stream>>>(att, afull, a1t, s);
    dim3 g3(N_ * 19, C_);
    k_agg<<<g3, 32, 0, stream>>>(x, a1t, aggT);
    dim3 g4(N_, 4, 47);
    k_z<<<g4, 32, 0, stream>>>(wdb, aggT, bd, wvec, out, s);
  }
  {
    size_t tot = (size_t)N_ * C_ * P_;
    k_bn<<<(unsigned)((tot + 255) / 256), 256, 0, stream>>>(out, x, gamma, beta, mean, var);
  }
}
